// MultiHeadAttention_33165737460082
// MI455X (gfx1250) — matrix operand order repaired and checked by reading, not compiled, not run on hardware
//
#include <hip/hip_runtime.h>
#include <hip/hip_bf16.h>

// ---------------------------------------------------------------------------
// MI455X (gfx1250) fused multi-head attention with RoPE, wave32 + WMMA f16.
// All matmuls use v_wmma_f32_16x16x32_f16. LDS tiles are laid out so every
// fragment load is an aligned ds_load_b128 (A: [m][k] stride-48; B: [n][k]).
// ---------------------------------------------------------------------------

typedef __attribute__((ext_vector_type(16))) _Float16 v16h;
typedef __attribute__((ext_vector_type(8)))  float    v8f;

union HFrag { v16h v; _Float16 e[16]; uint4 u[2]; };
union CFrag { v8f  v; float    e[8];  };
union H8    { uint4 u; _Float16 h[8]; };

#define NEG_INF (-__builtin_inff())

static constexpr int B_SZ   = 2;
static constexpr int S_LEN  = 1024;
static constexpr int D_MOD  = 1024;
static constexpr int H_CNT  = 16;
static constexpr int D_K    = 64;
static constexpr int M_ROWS = B_SZ * S_LEN;          // 2048
static constexpr size_t OUT_ELEMS = (size_t)B_SZ * S_LEN * D_MOD;            // 2,097,152
static constexpr size_t BHSS      = (size_t)B_SZ * H_CNT * S_LEN * S_LEN;    // 33,554,432

// LDS row stride in halves: 48 -> 96 bytes, multiple of 16B, conflict-skewed.
static constexpr int LSTR = 48;

// ---------------------------------------------------------------------------
// Kernel 1: generic GEMM  Y[m,n] = sum_k A[m,k] * W[n,k]   (A,W fp32)
// Block tile 64(M) x 128(N), K-step 32. 8 waves: wm in 0..3, wn in 0..1,
// each wave computes 16 x 64 via 4 WMMA accumulators.
// out_mode 0: head-split [B,H,S,64] f32; out_mode 1: flat [M,N] f32.
// ---------------------------------------------------------------------------
__global__ __launch_bounds__(256)
void gemm_xwT_kernel(const float* __restrict__ A, const float* __restrict__ W,
                     float* __restrict__ Y, int out_mode) {
  constexpr int K = 1024;
  __shared__ alignas(16) _Float16 lds_a[64][LSTR];    // [m][k]
  __shared__ alignas(16) _Float16 lds_bT[128][LSTR];  // [n][k]
  const int tid  = threadIdx.x;
  const int m0   = blockIdx.y * 64;
  const int n0   = blockIdx.x * 128;
  const int w    = tid >> 5, lane = tid & 31;
  const int wm   = w & 3,    wn   = w >> 2;
  const int half = lane >> 4, lr  = lane & 15;

  v8f acc[4] = {};

  for (int k0 = 0; k0 < K; k0 += 32) {
    // ---- stage A tile (64x32 f32 -> f16), one b128 store per thread ----
    {
      const int row = tid >> 2, cg = (tid & 3) * 8;
      const float4* p = reinterpret_cast<const float4*>(A + (size_t)(m0 + row) * K + k0 + cg);
      float4 f0 = p[0], f1 = p[1];
      H8 s;
      s.h[0] = (_Float16)f0.x; s.h[1] = (_Float16)f0.y;
      s.h[2] = (_Float16)f0.z; s.h[3] = (_Float16)f0.w;
      s.h[4] = (_Float16)f1.x; s.h[5] = (_Float16)f1.y;
      s.h[6] = (_Float16)f1.z; s.h[7] = (_Float16)f1.w;
      *reinterpret_cast<uint4*>(&lds_a[row][cg]) = s.u;
    }
    // ---- stage B tile transposed: lds_bT[nn][kk] = W[n0+nn, k0+kk] ----
    {
      const int nn = tid >> 1, kg = (tid & 1) * 16;
      const float4* p = reinterpret_cast<const float4*>(W + (size_t)(n0 + nn) * K + k0 + kg);
      float4 g0 = p[0], g1 = p[1], g2 = p[2], g3 = p[3];
      H8 t0, t1;
      t0.h[0] = (_Float16)g0.x; t0.h[1] = (_Float16)g0.y;
      t0.h[2] = (_Float16)g0.z; t0.h[3] = (_Float16)g0.w;
      t0.h[4] = (_Float16)g1.x; t0.h[5] = (_Float16)g1.y;
      t0.h[6] = (_Float16)g1.z; t0.h[7] = (_Float16)g1.w;
      t1.h[0] = (_Float16)g2.x; t1.h[1] = (_Float16)g2.y;
      t1.h[2] = (_Float16)g2.z; t1.h[3] = (_Float16)g2.w;
      t1.h[4] = (_Float16)g3.x; t1.h[5] = (_Float16)g3.y;
      t1.h[6] = (_Float16)g3.z; t1.h[7] = (_Float16)g3.w;
      *reinterpret_cast<uint4*>(&lds_bT[nn][kg])     = t0.u;
      *reinterpret_cast<uint4*>(&lds_bT[nn][kg + 8]) = t1.u;
    }
    if (k0 + 32 < K) {   // hint next tiles (global_prefetch_b8)
      __builtin_prefetch(A + (size_t)(m0 + (tid >> 2)) * K + k0 + 32, 0, 1);
      __builtin_prefetch(W + (size_t)(n0 + (tid >> 1)) * K + k0 + 32, 0, 1);
    }
    __syncthreads();

    // ---- fragments: two aligned ds_load_b128 each ----
    HFrag a;
    {
      const int row = wm * 16 + lr, kb = half * 16;
      a.u[0] = *reinterpret_cast<const uint4*>(&lds_a[row][kb]);
      a.u[1] = *reinterpret_cast<const uint4*>(&lds_a[row][kb + 8]);
    }
    #pragma unroll
    for (int sub = 0; sub < 4; ++sub) {
      HFrag b;
      const int col = wn * 64 + sub * 16 + lr, kb2 = half * 16;
      b.u[0] = *reinterpret_cast<const uint4*>(&lds_bT[col][kb2]);
      b.u[1] = *reinterpret_cast<const uint4*>(&lds_bT[col][kb2 + 8]);
      acc[sub] = __builtin_amdgcn_wmma_f32_16x16x32_f16(false, a.v, false, b.v,
                                                        (short)0, acc[sub], false, false);
    }
    __syncthreads();
  }

  #pragma unroll
  for (int sub = 0; sub < 4; ++sub) {
    CFrag cu; cu.v = acc[sub];
    const int ncol = n0 + wn * 64 + sub * 16 + lr;
    #pragma unroll
    for (int r = 0; r < 8; ++r) {
      const int m = m0 + wm * 16 + half * 8 + r;
      const int bb = m >> 10, s = m & 1023;
      if (out_mode == 0) {
        const int h = ncol >> 6, dd = ncol & 63;
        Y[(((size_t)(bb * H_CNT + h) * S_LEN) + s) * D_K + dd] = cu.e[r];
      } else {
        Y[(size_t)m * D_MOD + ncol] = cu.e[r];
      }
    }
  }
}

// ---------------------------------------------------------------------------
// Kernel 2: RoPE (Q,K) + f16 convert.  Q,K out: [B*H, S, 64].
// V out pre-transposed: Vt [B*H, 64, S]  (so attn B-tiles stage with b128).
// ---------------------------------------------------------------------------
__global__ __launch_bounds__(256)
void rope_cvt_kernel(const float* __restrict__ Qraw, const float* __restrict__ Kraw,
                     const float* __restrict__ Vraw,
                     _Float16* __restrict__ Qh, _Float16* __restrict__ Kh,
                     _Float16* __restrict__ Vt) {
  const int p = blockIdx.x * 256 + threadIdx.x;      // [0, B*H*S*32)
  const int dd2  = p & 31;
  const int rest = p >> 5;
  const int s    = rest & 1023;
  const int bh   = rest >> 10;
  const size_t idx = ((size_t)bh * S_LEN + s) * D_K + 2 * dd2;

  // theta = exp(-dd2 * ln(10000)/32), ang = (s+1)*theta
  const float theta = __expf(-(float)dd2 * 0.28782313662425576f);
  const float ang   = (float)(s + 1) * theta;
  float sn, cs; __sincosf(ang, &sn, &cs);

  const float q0 = Qraw[idx], q1 = Qraw[idx + 1];
  Qh[idx]     = (_Float16)(q0 * cs - q1 * sn);
  Qh[idx + 1] = (_Float16)(q1 * cs + q0 * sn);
  const float k0 = Kraw[idx], k1 = Kraw[idx + 1];
  Kh[idx]     = (_Float16)(k0 * cs - k1 * sn);
  Kh[idx + 1] = (_Float16)(k1 * cs + k0 * sn);

  const size_t vt = ((size_t)bh * D_K + 2 * dd2) * S_LEN + s;
  Vt[vt]         = (_Float16)Vraw[idx];
  Vt[vt + S_LEN] = (_Float16)Vraw[idx + 1];
}

// ---------------------------------------------------------------------------
// Kernel 3: per-head scores = (Q K^T) * 1/8 with q-pad / kv-pad / causal masks,
// written into prev_new[1] region of d_out.  M=N=1024, Kdim=64.
// Block tile 64 x 128, Kdim loop of 2.
// ---------------------------------------------------------------------------
__global__ __launch_bounds__(256)
void score_kernel(const _Float16* __restrict__ Qh, const _Float16* __restrict__ Kh,
                  const unsigned char* __restrict__ qmask,
                  const unsigned char* __restrict__ kvmask,
                  float* __restrict__ scores_out) {
  __shared__ alignas(16) _Float16 lds_a[64][LSTR];    // q rows  x k-dim
  __shared__ alignas(16) _Float16 lds_bT[128][LSTR];  // key cols x k-dim
  const int bh = blockIdx.z;
  const int bb = bh >> 4;
  const _Float16* Q  = Qh + (size_t)bh * S_LEN * D_K;
  const _Float16* Kp = Kh + (size_t)bh * S_LEN * D_K;
  const int tid  = threadIdx.x;
  const int m0   = blockIdx.y * 64;
  const int n0   = blockIdx.x * 128;
  const int w    = tid >> 5, lane = tid & 31;
  const int wm   = w & 3,    wn   = w >> 2;
  const int half = lane >> 4, lr  = lane & 15;

  v8f acc[4] = {};

  for (int k0 = 0; k0 < D_K; k0 += 32) {
    {
      const int row = tid >> 2, cg = (tid & 3) * 8;
      *reinterpret_cast<uint4*>(&lds_a[row][cg]) =
          *reinterpret_cast<const uint4*>(Q + (size_t)(m0 + row) * D_K + k0 + cg);
    }
    {
      const int nn = tid >> 1, dg = (tid & 1) * 16;
      const uint4* p = reinterpret_cast<const uint4*>(Kp + (size_t)(n0 + nn) * D_K + k0 + dg);
      *reinterpret_cast<uint4*>(&lds_bT[nn][dg])     = p[0];
      *reinterpret_cast<uint4*>(&lds_bT[nn][dg + 8]) = p[1];
    }
    __syncthreads();

    HFrag a;
    {
      const int row = wm * 16 + lr, kb = half * 16;
      a.u[0] = *reinterpret_cast<const uint4*>(&lds_a[row][kb]);
      a.u[1] = *reinterpret_cast<const uint4*>(&lds_a[row][kb + 8]);
    }
    #pragma unroll
    for (int sub = 0; sub < 4; ++sub) {
      HFrag b;
      const int col = wn * 64 + sub * 16 + lr, kb2 = half * 16;
      b.u[0] = *reinterpret_cast<const uint4*>(&lds_bT[col][kb2]);
      b.u[1] = *reinterpret_cast<const uint4*>(&lds_bT[col][kb2 + 8]);
      acc[sub] = __builtin_amdgcn_wmma_f32_16x16x32_f16(false, a.v, false, b.v,
                                                        (short)0, acc[sub], false, false);
    }
    __syncthreads();
  }

  #pragma unroll
  for (int sub = 0; sub < 4; ++sub) {
    CFrag cu; cu.v = acc[sub];
    const int kc = n0 + wn * 64 + sub * 16 + lr;
    const bool kp = kvmask[bb * S_LEN + kc] != 0;
    #pragma unroll
    for (int r = 0; r < 8; ++r) {
      const int q = m0 + wm * 16 + half * 8 + r;
      const bool masked = (qmask[bb * S_LEN + q] != 0) || kp || (kc > q);
      scores_out[((size_t)bh * S_LEN + q) * S_LEN + kc] =
          masked ? NEG_INF : cu.e[r] * 0.125f;
    }
  }
}

// ---------------------------------------------------------------------------
// Kernel 4: row softmax of avg = 0.5*(prev + scores); padded q rows -> 0.
// One block (256 threads) per row; output f16 weights.
// ---------------------------------------------------------------------------
__global__ __launch_bounds__(256)
void softmax_kernel(const float* __restrict__ prev, const float* __restrict__ scores,
                    const unsigned char* __restrict__ qmask,
                    _Float16* __restrict__ Wsoft) {
  __shared__ float red[256];
  const int r  = blockIdx.x;             // [0, B*H*S)
  const int bh = r >> 10, q = r & 1023, bb = bh >> 4;
  const int t  = threadIdx.x;
  const size_t base = (size_t)r * S_LEN;

  if (qmask[bb * S_LEN + q] != 0) {
    #pragma unroll
    for (int j = 0; j < 4; ++j) Wsoft[base + t + j * 256] = (_Float16)0.0f;
    return;
  }

  float vals[4];
  float mx = NEG_INF;
  #pragma unroll
  for (int j = 0; j < 4; ++j) {
    const float v = 0.5f * (prev[base + t + j * 256] + scores[base + t + j * 256]);
    vals[j] = v; mx = fmaxf(mx, v);
  }
  red[t] = mx; __syncthreads();
  for (int off = 128; off > 0; off >>= 1) {
    if (t < off) red[t] = fmaxf(red[t], red[t + off]);
    __syncthreads();
  }
  mx = red[0]; __syncthreads();

  float sum = 0.0f;
  #pragma unroll
  for (int j = 0; j < 4; ++j) { const float e = __expf(vals[j] - mx); vals[j] = e; sum += e; }
  red[t] = sum; __syncthreads();
  for (int off = 128; off > 0; off >>= 1) {
    if (t < off) red[t] += red[t + off];
    __syncthreads();
  }
  const float inv = 1.0f / red[0];
  #pragma unroll
  for (int j = 0; j < 4; ++j) Wsoft[base + t + j * 256] = (_Float16)(vals[j] * inv);
}

// ---------------------------------------------------------------------------
// Kernel 5: per-head attn = W @ V.  M=1024, N=64, K=1024.
// V comes pre-transposed (Vt [64][S]) so B stages as contiguous b128.
// Output [B,S,D] f32 (head-merged) for the final projection.
// ---------------------------------------------------------------------------
__global__ __launch_bounds__(256)
void attn_kernel(const _Float16* __restrict__ Wsoft, const _Float16* __restrict__ Vt,
                 float* __restrict__ attn) {
  __shared__ alignas(16) _Float16 lds_a[64][LSTR];   // q rows x k
  __shared__ alignas(16) _Float16 lds_bT[64][LSTR];  // d cols x k
  const int bh = blockIdx.z;
  const int bb = bh >> 4, h = bh & 15;
  const _Float16* Wp = Wsoft + (size_t)bh * S_LEN * S_LEN;
  const _Float16* V  = Vt    + (size_t)bh * D_K * S_LEN;
  const int tid  = threadIdx.x;
  const int m0   = blockIdx.y * 64;
  const int w    = tid >> 5, lane = tid & 31;
  const int wm   = w & 3,    wn   = w >> 2;
  const int half = lane >> 4, lr  = lane & 15;

  v8f acc0 = {}; v8f acc1 = {};

  for (int k0 = 0; k0 < S_LEN; k0 += 32) {
    {
      const int row = tid >> 2, cg = (tid & 3) * 8;
      *reinterpret_cast<uint4*>(&lds_a[row][cg]) =
          *reinterpret_cast<const uint4*>(Wp + (size_t)(m0 + row) * S_LEN + k0 + cg);
    }
    {
      const int dd = tid >> 2, kg = (tid & 3) * 8;
      *reinterpret_cast<uint4*>(&lds_bT[dd][kg]) =
          *reinterpret_cast<const uint4*>(V + (size_t)dd * S_LEN + k0 + kg);
    }
    if (k0 + 32 < S_LEN) {
      __builtin_prefetch(Wp + (size_t)(m0 + (tid >> 2)) * S_LEN + k0 + 32, 0, 1);
    }
    __syncthreads();

    HFrag a;
    {
      const int row = wm * 16 + lr, kb = half * 16;
      a.u[0] = *reinterpret_cast<const uint4*>(&lds_a[row][kb]);
      a.u[1] = *reinterpret_cast<const uint4*>(&lds_a[row][kb + 8]);
    }
    HFrag b0, b1;
    {
      const int kb2 = half * 16;
      const int c0 = wn * 32 + lr, c1 = wn * 32 + 16 + lr;
      b0.u[0] = *reinterpret_cast<const uint4*>(&lds_bT[c0][kb2]);
      b0.u[1] = *reinterpret_cast<const uint4*>(&lds_bT[c0][kb2 + 8]);
      b1.u[0] = *reinterpret_cast<const uint4*>(&lds_bT[c1][kb2]);
      b1.u[1] = *reinterpret_cast<const uint4*>(&lds_bT[c1][kb2 + 8]);
    }
    acc0 = __builtin_amdgcn_wmma_f32_16x16x32_f16(false, a.v, false, b0.v, (short)0, acc0, false, false);
    acc1 = __builtin_amdgcn_wmma_f32_16x16x32_f16(false, a.v, false, b1.v, (short)0, acc1, false, false);
    __syncthreads();
  }

  CFrag c0u, c1u; c0u.v = acc0; c1u.v = acc1;
  #pragma unroll
  for (int r = 0; r < 8; ++r) {
    const int q = m0 + wm * 16 + half * 8 + r;
    #pragma unroll
    for (int sub = 0; sub < 2; ++sub) {
      const int dcol = wn * 32 + sub * 16 + lr;       // N = 64, single block in x
      const float val = sub ? c1u.e[r] : c0u.e[r];
      attn[((size_t)(bb * S_LEN + q)) * D_MOD + h * D_K + dcol] = val;
    }
  }
}

// ---------------------------------------------------------------------------
// Host-side launch sequence (graph-capture safe: only kernels + async D2D).
// ---------------------------------------------------------------------------
extern "C" void kernel_launch(void* const* d_in, const int* in_sizes, int n_in,
                              void* d_out, int out_size, void* d_ws, size_t ws_size,
                              hipStream_t stream) {
  const float* src_q   = (const float*)d_in[0];
  const float* src_kv  = (const float*)d_in[1];
  const unsigned char* qmask  = (const unsigned char*)d_in[2];
  const unsigned char* kvmask = (const unsigned char*)d_in[3];
  const float* prev    = (const float*)d_in[4];
  const float* Wq      = (const float*)d_in[5];
  const float* Wk      = (const float*)d_in[6];
  const float* Wv      = (const float*)d_in[7];
  const float* Wproj   = (const float*)d_in[8];
  float* out = (float*)d_out;

  // Workspace layout (raw Q/K/V regions are dead before Wsoft is written):
  //   [0,12MB)    : Qh, Kh, Vt (f16)
  //   [12MB,...)  : Qraw/Kraw/Vraw (f32, temp)  -- later reused by Wsoft (f16)
  //   [76MB,84MB) : attn (f32)
  char* ws = (char*)d_ws;
  _Float16* Qh  = (_Float16*)(ws + 0);
  _Float16* Kh  = (_Float16*)(ws + 4194304);
  _Float16* Vt  = (_Float16*)(ws + 8388608);
  float* Qraw   = (float*)(ws + 12582912);
  float* Kraw   = (float*)(ws + 20971520);
  float* Vraw   = (float*)(ws + 29360128);
  _Float16* Wsoft = (_Float16*)(ws + 12582912);   // reuses raw region
  float* attn   = (float*)(ws + 79691776);

  float* prev0_dst  = out + OUT_ELEMS;            // prev_new[0]
  float* scores_dst = out + OUT_ELEMS + BHSS;     // prev_new[1]

  // prev_new[0] = prev (async device copy, capture-safe)
  hipMemcpyAsync(prev0_dst, prev, BHSS * sizeof(float),
                 hipMemcpyDeviceToDevice, stream);

  const dim3 blk(256);
  const dim3 grid_gemm(D_MOD / 128, M_ROWS / 64);          // (8, 32)

  // Q/K/V projections (head-split f32 output)
  gemm_xwT_kernel<<<grid_gemm, blk, 0, stream>>>(src_q,  Wq, Qraw, 0);
  gemm_xwT_kernel<<<grid_gemm, blk, 0, stream>>>(src_kv, Wk, Kraw, 0);
  gemm_xwT_kernel<<<grid_gemm, blk, 0, stream>>>(src_kv, Wv, Vraw, 0);

  // RoPE + f16 convert (V pre-transposed):  B*H*S*32 pair-threads
  rope_cvt_kernel<<<dim3((B_SZ * H_CNT * S_LEN * 32) / 256), blk, 0, stream>>>(
      Qraw, Kraw, Vraw, Qh, Kh, Vt);

  // Masked, scaled scores -> prev_new[1]
  score_kernel<<<dim3(S_LEN / 128, S_LEN / 64, B_SZ * H_CNT), blk, 0, stream>>>(
      Qh, Kh, qmask, kvmask, scores_dst);

  // Softmax of 0.5*(prev + scores) -> f16 weights
  softmax_kernel<<<dim3(B_SZ * H_CNT * S_LEN), blk, 0, stream>>>(
      prev, scores_dst, qmask, Wsoft);

  // attn = W @ V  -> [B,S,D] f32
  attn_kernel<<<dim3(1, S_LEN / 64, B_SZ * H_CNT), blk, 0, stream>>>(Wsoft, Vt, attn);

  // Final projection: out = attn @ Wproj^T (flat output into d_out)
  gemm_xwT_kernel<<<grid_gemm, blk, 0, stream>>>(attn, Wproj, out, 1);
}
